// RNN_50036368998466
// MI455X (gfx1250) — compile-verified, matrix operand order checked by reading, not run on hardware
//
#include <hip/hip_runtime.h>

typedef __attribute__((ext_vector_type(8)))  _Float16 v8h;
typedef __attribute__((ext_vector_type(16))) _Float16 v16h;
typedef __attribute__((ext_vector_type(8)))  float    v8f;

#define WAVES 8
#define TILES_PER_WAVE 8

__device__ __forceinline__ void lds_fence_wave() {
  // Same-wave LDS ops are processed in order; this wait + memory clobber keeps
  // the compiler from reordering cross-lane LDS producer/consumer pairs.
  asm volatile("s_wait_dscnt 0" ::: "memory");
}

// Branch-free tanh: prefer the CDNA5 V_TANH_F32 transcendental, else compose
// from v_exp_f32 / v_rcp_f32 (saturates correctly at +-1 via inf/0, no EXEC
// divergence -- unlike ocml tanhf which emits v_cmpx + branches per element).
__device__ __forceinline__ float fast_tanh(float x) {
#if __has_builtin(__builtin_amdgcn_tanhf)
  return __builtin_amdgcn_tanhf(x);
#else
  float e = __builtin_amdgcn_exp2f(x * 2.885390081777927f);  // exp(2x)
  return 1.0f - 2.0f * __builtin_amdgcn_rcpf(e + 1.0f);
#endif
}

// Load a 16x32 (f16) A-tile row chunk / B-tile column chunk per the CDNA5
// 16-bit WMMA VGPR layout: two 16-byte pieces at elemOff and elemOff+16.
__device__ __forceinline__ v16h ld_tile16(const _Float16* base, int elemOff) {
  v8h lo = *(const v8h*)(base + elemOff);
  v8h hi = *(const v8h*)(base + elemOff + 16);
  return __builtin_shufflevector(lo, hi, 0,1,2,3,4,5,6,7,8,9,10,11,12,13,14,15);
}

__device__ __forceinline__ v8f wmma_f16(v16h a, v16h b, v8f c) {
  return __builtin_amdgcn_wmma_f32_16x16x32_f16(false, a, false, b, (short)0, c,
                                                false, false);
}

__global__ __launch_bounds__(256) void rnn_mlp_fused(
    const float* __restrict__ xg,
    const float* __restrict__ W_ih, const float* __restrict__ W_hh,
    const float* __restrict__ b_ih, const float* __restrict__ b_hh,
    const float* __restrict__ W1,   const float* __restrict__ b1,
    const float* __restrict__ W2,   const float* __restrict__ b2,
    const float* __restrict__ W3,   const float* __restrict__ b3,
    float* __restrict__ out, int Btotal)
{
  // B-matrix LDS images, stored column-contiguous over K so each lane's column
  // chunk is two contiguous 16B pieces. B[k][n] = W[n][k] => identity copy of
  // the row-major weight.
  __shared__ _Float16 sWhh[32*32] __attribute__((aligned(16)));
  __shared__ _Float16 sW1 [16*96] __attribute__((aligned(16)));
  __shared__ _Float16 sW2p[16*32] __attribute__((aligned(16)));   // zero padded K->32, N->16
  __shared__ float sWih[32], sBsum[32], sB1[16], sB2[16], sW3v[8], sB3[1];
  // Per-wave staging buffers (row-major f16 so A loads are 2x ds_load_b128).
  __shared__ _Float16 hA [WAVES][16*32] __attribute__((aligned(16)));
  __shared__ _Float16 fcA[WAVES][16*96] __attribute__((aligned(16)));
  __shared__ _Float16 a2 [WAVES][16*32] __attribute__((aligned(16)));
  __shared__ float xs [WAVES][64];
  __shared__ float oSt[WAVES][16*16];

  const int tid = threadIdx.x;

  for (int i = tid; i < 32*32; i += 256) sWhh[i] = (_Float16)W_hh[i];
  for (int i = tid; i < 16*96; i += 256) sW1[i]  = (_Float16)W1[i];
  for (int i = tid; i < 16*32; i += 256) {
    int nn = i >> 5, kk = i & 31;
    sW2p[i] = (nn < 8 && kk < 16) ? (_Float16)W2[nn*16 + kk] : (_Float16)0.0f;
  }
  for (int i = tid; i < WAVES*16*32; i += 256) ((_Float16*)a2)[i] = (_Float16)0.0f;
  if (tid < 32)       { sWih[tid] = W_ih[tid]; sBsum[tid] = b_ih[tid] + b_hh[tid]; }
  else if (tid < 48)  { sB1[tid-32] = b1[tid-32]; }
  else if (tid < 64)  { sB2[tid-48] = (tid-48 < 8) ? b2[tid-48] : 0.0f; }
  else if (tid < 72)  { sW3v[tid-64] = W3[tid-64]; }
  else if (tid == 72) { sB3[0] = b3[0]; }
  __syncthreads();

  const int wave = tid >> 5;
  const int lane = tid & 31;
  const int n    = lane & 15;   // column (B/C) or row (A) owned by this lane
  const int hi   = lane >> 4;   // K-half / row-half selector

  _Float16* hA_w  = hA[wave];
  _Float16* fcA_w = fcA[wave];
  _Float16* a2_w  = a2[wave];
  float*    xsw   = xs[wave];
  float*    oSt_w = oSt[wave];

  // B-matrix register tiles, held across the whole tile loop.
  v16h Bhh0  = ld_tile16(sWhh, (n     )*32 + 8*hi);
  v16h Bhh1  = ld_tile16(sWhh, (n + 16)*32 + 8*hi);
  v16h BW1c0 = ld_tile16(sW1,  n*96 +  0 + 8*hi);
  v16h BW1c1 = ld_tile16(sW1,  n*96 + 32 + 8*hi);
  v16h BW1c2 = ld_tile16(sW1,  n*96 + 64 + 8*hi);
  v16h BW2   = ld_tile16(sW2p, n*32 + 8*hi);

  const float wih0 = sWih[n],  wih1 = sWih[n + 16];
  const float bs0  = sBsum[n], bs1  = sBsum[n + 16];
  const float b1v  = sB1[n];
  const float b2v  = sB2[n];
  float w3r[8];
  #pragma unroll
  for (int i = 0; i < 8; ++i) w3r[i] = sW3v[i];
  const float b3v = sB3[0];

  // step-1 per-lane vectors: lane handles row n, columns hi*16 .. hi*16+15
  float wihc[16], bsc[16];
  #pragma unroll
  for (int i = 0; i < 16; ++i) { wihc[i] = sWih[hi*16 + i]; bsc[i] = sBsum[hi*16 + i]; }

  const long tile0 = ((long)blockIdx.x * WAVES + wave) * TILES_PER_WAVE;
  for (int it = 0; it < TILES_PER_WAVE; ++it) {
    const long R = (tile0 + it) * 16;
    if (R >= (long)Btotal) break;

    // ---- stage x[R..R+15][0..2] (48 floats) into LDS ----
    const long xbase = R * 3;
    float xv0 = (xbase + lane      < (long)Btotal*3) ? xg[xbase + lane]      : 0.0f;
    float xv1 = 0.0f;
    if (lane < 16 && xbase + 32 + lane < (long)Btotal*3) xv1 = xg[xbase + 32 + lane];
    xsw[lane] = xv0;
    if (lane < 16) xsw[32 + lane] = xv1;
    if (it + 1 < TILES_PER_WAVE) __builtin_prefetch(&xg[xbase + 48], 0, 1);
    lds_fence_wave();

    // ---- step 1: h1 = tanh(x0*W_ih + b_ih + b_hh)   (h0 == 0, no matmul) ----
    {
      const float xv = xsw[n*3 + 0];
      v8h p0, p1, q0, q1;
      #pragma unroll
      for (int i = 0; i < 8; ++i) {
        float h0v = fast_tanh(fmaf(xv, wihc[i],     bsc[i]));
        float h1v = fast_tanh(fmaf(xv, wihc[i + 8], bsc[i + 8]));
        p0[i] = (_Float16)h0v;              p1[i] = (_Float16)h1v;
        q0[i] = (_Float16)fmaxf(h0v, 0.f);  q1[i] = (_Float16)fmaxf(h1v, 0.f);
      }
      *(v8h*)&hA_w [n*32 + hi*16]     = p0;
      *(v8h*)&hA_w [n*32 + hi*16 + 8] = p1;
      *(v8h*)&fcA_w[n*96 + hi*16]     = q0;
      *(v8h*)&fcA_w[n*96 + hi*16 + 8] = q1;
    }

    // ---- steps 2..3: h_t = tanh(h_{t-1} @ W_hh^T + x_t*W_ih + b)  via WMMA ----
    #pragma unroll
    for (int t = 1; t < 3; ++t) {
      lds_fence_wave();
      v16h A = ld_tile16(hA_w, n*32 + 8*hi);
      v8f P0, P1;                       // bias + x*W_ih folded into C input
      #pragma unroll
      for (int r = 0; r < 8; ++r) {
        float xv = xsw[(r + 8*hi)*3 + t];
        P0[r] = fmaf(xv, wih0, bs0);
        P1[r] = fmaf(xv, wih1, bs1);
      }
      v8f C0 = wmma_f16(A, Bhh0, P0);
      v8f C1 = wmma_f16(A, Bhh1, P1);
      #pragma unroll
      for (int r = 0; r < 8; ++r) {
        const int row = r + 8*hi;
        float h0v = fast_tanh(C0[r]);
        float h1v = fast_tanh(C1[r]);
        if (t < 2) {
          hA_w[row*32 + n]      = (_Float16)h0v;
          hA_w[row*32 + n + 16] = (_Float16)h1v;
        }
        fcA_w[row*96 + t*32 + n]      = (_Float16)fmaxf(h0v, 0.f);
        fcA_w[row*96 + t*32 + n + 16] = (_Float16)fmaxf(h1v, 0.f);
      }
    }

    // ---- fc1: [16,96] @ W1^T + b1 -> [16,16]  (K=96 = 3 accumulating WMMAs) ----
    lds_fence_wave();
    v8f O1;
    #pragma unroll
    for (int r = 0; r < 8; ++r) O1[r] = b1v;
    O1 = wmma_f16(ld_tile16(fcA_w, n*96 +  0 + 8*hi), BW1c0, O1);
    O1 = wmma_f16(ld_tile16(fcA_w, n*96 + 32 + 8*hi), BW1c1, O1);
    O1 = wmma_f16(ld_tile16(fcA_w, n*96 + 64 + 8*hi), BW1c2, O1);

    // stage o1 as fc2 A-matrix (K cols 16..31 stay zero from init)
    #pragma unroll
    for (int r = 0; r < 8; ++r) a2_w[(r + 8*hi)*32 + n] = (_Float16)O1[r];
    lds_fence_wave();

    // ---- fc2: relu([16,16] @ W2^T + b2) -> [16,8] (zero-padded WMMA) ----
    v8f P2;
    #pragma unroll
    for (int r = 0; r < 8; ++r) P2[r] = b2v;
    v8f O2 = wmma_f16(ld_tile16(a2_w, n*32 + 8*hi), BW2, P2);
    #pragma unroll
    for (int r = 0; r < 8; ++r) oSt_w[(r + 8*hi)*16 + n] = fmaxf(O2[r], 0.f);
    lds_fence_wave();

    // ---- fc3: [16,8] @ W3^T + b3 -> [16,1] ----
    if (lane < 16 && R + lane < (long)Btotal) {
      const float* orow = &oSt_w[lane*16];
      float acc = b3v;
      #pragma unroll
      for (int i = 0; i < 8; ++i) acc = fmaf(orow[i], w3r[i], acc);
      out[R + lane] = acc;
    }
  }
}

extern "C" void kernel_launch(void* const* d_in, const int* in_sizes, int n_in,
                              void* d_out, int out_size, void* d_ws, size_t ws_size,
                              hipStream_t stream) {
  const float* x    = (const float*)d_in[0];
  const float* W_ih = (const float*)d_in[1];
  const float* W_hh = (const float*)d_in[2];
  const float* b_ih = (const float*)d_in[3];
  const float* b_hh = (const float*)d_in[4];
  const float* W1   = (const float*)d_in[5];
  const float* b1   = (const float*)d_in[6];
  const float* W2   = (const float*)d_in[7];
  const float* b2   = (const float*)d_in[8];
  const float* W3   = (const float*)d_in[9];
  const float* b3   = (const float*)d_in[10];
  float* out = (float*)d_out;

  const int Btotal = in_sizes[0] / 3;              // x is [B, 3, 1]
  const int rowsPerBlock = WAVES * TILES_PER_WAVE * 16;  // 1024
  const int grid = (Btotal + rowsPerBlock - 1) / rowsPerBlock;

  hipLaunchKernelGGL(rnn_mlp_fused, dim3(grid), dim3(256), 0, stream,
                     x, W_ih, W_hh, b_ih, b_hh, W1, b1, W2, b2, W3, b3,
                     out, Btotal);
}
